// WorkingMemoryModule_78597901517448
// MI455X (gfx1250) — matrix operand order repaired and checked by reading, not compiled, not run on hardware
//
#include <hip/hip_runtime.h>

// ---------------------------------------------------------------------------
// Types for CDNA5 WMMA (wave32): v_wmma_f32_16x16x32_bf16
// ---------------------------------------------------------------------------
typedef __attribute__((ext_vector_type(16))) __bf16 v16bf;
typedef __attribute__((ext_vector_type(8)))  __bf16 v8bf;
typedef __attribute__((ext_vector_type(4)))  __bf16 v4bf;
typedef __attribute__((ext_vector_type(8)))  float  v8f;

#define B_BATCH 64
#define H_DIM   4096

// ---------------------------------------------------------------------------
// A-fragment loader: weights W[n,k] row-major, fp32 -> bf16 in registers.
// 16-bit A 16x32 layout: lane L<16 holds M=L, K={0..7,16..23}; lane L+16 holds
// M=L, K={8..15,24..31}. Caller passes p = W + row*K + 8*laneHi + k0, so this
// loads K-chunks [0..7] and [16..23] relative to p. Lane pairs (L, L+16)
// together consume a full contiguous 128B row segment per K-step.
// ---------------------------------------------------------------------------
__device__ __forceinline__ v16bf load_wfrag(const float* __restrict__ p) {
    float4 w0 = *(const float4*)(p);
    float4 w1 = *(const float4*)(p + 4);
    float4 w2 = *(const float4*)(p + 16);
    float4 w3 = *(const float4*)(p + 20);
    v16bf a;
    a[0]  = (__bf16)w0.x; a[1]  = (__bf16)w0.y; a[2]  = (__bf16)w0.z; a[3]  = (__bf16)w0.w;
    a[4]  = (__bf16)w1.x; a[5]  = (__bf16)w1.y; a[6]  = (__bf16)w1.z; a[7]  = (__bf16)w1.w;
    a[8]  = (__bf16)w2.x; a[9]  = (__bf16)w2.y; a[10] = (__bf16)w2.z; a[11] = (__bf16)w2.w;
    a[12] = (__bf16)w3.x; a[13] = (__bf16)w3.y; a[14] = (__bf16)w3.z; a[15] = (__bf16)w3.w;
    return a;
}

// ---------------------------------------------------------------------------
// B-fragment loader: activations act[b,k] row-major bf16. B 32x16 layout:
// lanes 0-15 hold K={0..7,16..23} (N=lane), lanes 16-31 hold K={8..15,24..31} (N=lane-16).
// Caller passes p = act + (b0+laneLo)*K + 8*laneHi + k0 -> bf16 chunks [0..7] and [16..23].
// ---------------------------------------------------------------------------
__device__ __forceinline__ v16bf load_bfrag(const __bf16* __restrict__ p) {
    v8bf lo = *(const v8bf*)(p);
    v8bf hi = *(const v8bf*)(p + 16);
    v16bf b;
#pragma unroll
    for (int j = 0; j < 8; ++j) { b[j] = lo[j]; b[8 + j] = hi[j]; }
    return b;
}

// ---------------------------------------------------------------------------
// Fused (dual) GEMM: outT[n, b] = sum_k W1[n,k]*act1[b,k] (+ W2[n,k]*act2[b,k])
//                               + bias1[n] (+ bias2[n]) , optional ReLU.
// M = 64 (batch), Ntot = rows of W, K = 4096. Weights converted fp32->bf16 in
// registers; each weight byte read from HBM exactly once (bandwidth floor).
// Block = 256 threads (8 waves); each wave: 16 n-rows x full batch 64.
// ---------------------------------------------------------------------------
template <bool DUAL, bool RELU>
__global__ __launch_bounds__(256)
void gemm_gates_kernel(const float* __restrict__ W1, const float* __restrict__ W2,
                       const __bf16* __restrict__ act1, const __bf16* __restrict__ act2,
                       const float* __restrict__ bias1, const float* __restrict__ bias2,
                       float* __restrict__ outT, int K) {
    const int wave   = threadIdx.x >> 5;
    const int lane   = threadIdx.x & 31;
    const int laneLo = lane & 15;
    const int laneHi = lane >> 4;             // 0 or 1
    const int n0     = blockIdx.x * 128 + wave * 16;
    const int row    = n0 + laneLo;           // weight row this lane serves

    const float* w1p = W1 + (size_t)row * K + 8 * laneHi;
    const float* w2p = DUAL ? (W2 + (size_t)row * K + 8 * laneHi) : nullptr;

    const __bf16* b1p[4];
    const __bf16* b2p[4];
#pragma unroll
    for (int t = 0; t < 4; ++t) {
        b1p[t] = act1 + (size_t)(16 * t + laneLo) * K + 8 * laneHi;
        if (DUAL) b2p[t] = act2 + (size_t)(16 * t + laneLo) * K + 8 * laneHi;
    }

    v8f acc[4] = {};

#pragma unroll 2
    for (int k0 = 0; k0 < K; k0 += 32) {
        v16bf a1 = load_wfrag(w1p + k0);
#pragma unroll
        for (int t = 0; t < 4; ++t) {
            v16bf b1 = load_bfrag(b1p[t] + k0);
            acc[t] = __builtin_amdgcn_wmma_f32_16x16x32_bf16(
                false, a1, false, b1, (short)0, acc[t], false, false);
        }
        if constexpr (DUAL) {
            v16bf a2 = load_wfrag(w2p + k0);
#pragma unroll
            for (int t = 0; t < 4; ++t) {
                v16bf b2 = load_bfrag(b2p[t] + k0);
                acc[t] = __builtin_amdgcn_wmma_f32_16x16x32_bf16(
                    false, a2, false, b2, (short)0, acc[t], false, false);
            }
        }
    }

    // D layout: element r of v8f -> n = n0 + r + 8*laneHi, batch col = laneLo.
    float bsum[8];
#pragma unroll
    for (int r = 0; r < 8; ++r) {
        const int n = n0 + r + 8 * laneHi;
        float bb = bias1[n];
        if constexpr (DUAL) bb += bias2[n];
        bsum[r] = bb;
    }
#pragma unroll
    for (int t = 0; t < 4; ++t) {
#pragma unroll
        for (int r = 0; r < 8; ++r) {
            float v = acc[t][r] + bsum[r];
            if constexpr (RELU) v = fmaxf(v, 0.0f);
            const int n = n0 + r + 8 * laneHi;
            outT[(size_t)n * B_BATCH + 16 * t + laneLo] = v;
        }
    }
}

// ---------------------------------------------------------------------------
// Pointwise LSTM cell from transposed gates gatesT[4H, B].
// gates are [n,b]-major but h/c are [b,n]-major, so a 64n x 64b tile is
// transposed through padded LDS (stride 65 -> bank-conflict-free):
//   phase 1: coalesced gate reads -> sigmoids/tanh -> (f, i*g, o) into LDS
//   phase 2: transposed LDS reads -> coalesced c read + h/c/bf16 writes.
// Every global access in this kernel is fully coalesced.
// ---------------------------------------------------------------------------
__global__ __launch_bounds__(256)
void lstm_pointwise_kernel(const float* __restrict__ gatesT,
                           const float* __restrict__ c_prev,
                           float* __restrict__ h_out, float* __restrict__ c_out,
                           float* __restrict__ h_dup, __bf16* __restrict__ h_bf) {
    __shared__ float s_f [64 * 65];
    __shared__ float s_ig[64 * 65];
    __shared__ float s_o [64 * 65];

    const int tid = threadIdx.x;
    const int n0  = blockIdx.x * 64;
    const size_t HB   = (size_t)H_DIM * B_BATCH;
    const size_t base = (size_t)n0 * B_BATCH;

#pragma unroll
    for (int j = 0; j < 16; ++j) {
        const int l  = j * 256 + tid;          // 0..4095 over the tile
        const int nl = l >> 6;                 // local n
        const int b  = l & 63;                 // batch (fast -> coalesced)
        const size_t g = base + (size_t)l;
        const float gi = gatesT[g];
        const float gf = gatesT[g + HB];
        const float gg = gatesT[g + 2 * HB];
        const float go = gatesT[g + 3 * HB];
        const float i_ = 1.0f / (1.0f + __expf(-gi));
        const float f_ = 1.0f / (1.0f + __expf(-gf));
        const float g_ = tanhf(gg);
        const float o_ = 1.0f / (1.0f + __expf(-go));
        s_f [nl * 65 + b] = f_;
        s_ig[nl * 65 + b] = i_ * g_;
        s_o [nl * 65 + b] = o_;
    }
    __syncthreads();
#pragma unroll
    for (int j = 0; j < 16; ++j) {
        const int l = j * 256 + tid;
        const int b = l >> 6;                  // batch row
        const int n = l & 63;                  // local n (fast -> coalesced)
        const float f_ = s_f [n * 65 + b];
        const float ig = s_ig[n * 65 + b];
        const float o_ = s_o [n * 65 + b];
        const size_t oi = (size_t)b * H_DIM + n0 + n;
        const float cn = f_ * c_prev[oi] + ig;
        const float hn = o_ * tanhf(cn);
        c_out[oi] = cn;
        h_out[oi] = hn;
        if (h_dup) h_dup[oi] = hn;
        h_bf[oi] = (__bf16)hn;
    }
}

// ---------------------------------------------------------------------------
// Count head final layer: out[b] = relu( sum_n hidT[n,b] * Wc2[n] + bc2 ).
// 256 threads: 4-way split-K per batch element + LDS reduction (deterministic).
// ---------------------------------------------------------------------------
__global__ __launch_bounds__(256)
void count_head_kernel(const float* __restrict__ hidT,
                       const float* __restrict__ Wc2,
                       const float* __restrict__ bc2,
                       float* __restrict__ out) {
    __shared__ float red[256];
    const int b = threadIdx.x & 63;
    const int p = threadIdx.x >> 6;            // 0..3
    float s = 0.0f;
#pragma unroll 4
    for (int n = p; n < H_DIM / 2; n += 4)
        s += hidT[(size_t)n * B_BATCH + b] * Wc2[n];
    red[threadIdx.x] = s;
    __syncthreads();
    if (threadIdx.x < 64) {
        const float t = red[b] + red[64 + b] + red[128 + b] + red[192 + b];
        out[b] = fmaxf(t + bc2[0], 0.0f);
    }
}

// ---------------------------------------------------------------------------
// fp32 -> bf16 conversion (vectorized x4) for the small activation tensors.
// ---------------------------------------------------------------------------
__global__ __launch_bounds__(256)
void f32_to_bf16_kernel(const float* __restrict__ in, __bf16* __restrict__ out, int n4) {
    const int i = blockIdx.x * blockDim.x + threadIdx.x;
    if (i < n4) {
        float4 v = ((const float4*)in)[i];
        v4bf o;
        o[0] = (__bf16)v.x; o[1] = (__bf16)v.y; o[2] = (__bf16)v.z; o[3] = (__bf16)v.w;
        ((v4bf*)out)[i] = o;
    }
}

// ---------------------------------------------------------------------------
// Host-side orchestration
// ---------------------------------------------------------------------------
extern "C" void kernel_launch(void* const* d_in, const int* in_sizes, int n_in,
                              void* d_out, int out_size, void* d_ws, size_t ws_size,
                              hipStream_t stream) {
    (void)in_sizes; (void)n_in; (void)out_size; (void)ws_size;

    const float* x     = (const float*)d_in[0];   // [64, 4096]
    const float* h0    = (const float*)d_in[1];   // [2, 64, 4096]
    const float* c0    = (const float*)d_in[2];   // [2, 64, 4096]
    const float* W_ih0 = (const float*)d_in[3];
    const float* W_hh0 = (const float*)d_in[4];
    const float* b_ih0 = (const float*)d_in[5];
    const float* b_hh0 = (const float*)d_in[6];
    const float* W_ih1 = (const float*)d_in[7];
    const float* W_hh1 = (const float*)d_in[8];
    const float* b_ih1 = (const float*)d_in[9];
    const float* b_hh1 = (const float*)d_in[10];
    const float* Wc1   = (const float*)d_in[11];  // [2048, 4096]
    const float* bc1   = (const float*)d_in[12];
    const float* Wc2   = (const float*)d_in[13];  // [1, 2048]
    const float* bc2   = (const float*)d_in[14];

    const int BH = B_BATCH * H_DIM;               // 262144

    // d_out layout: memory_state | h_new[2,B,H] | c_new[2,B,H] | count[64]
    float* out_ms  = (float*)d_out;
    float* out_h   = out_ms + BH;
    float* out_c   = out_h + 2 * BH;
    float* out_cnt = out_c + 2 * BH;

    // Workspace layout (bytes)
    char* ws = (char*)d_ws;
    __bf16* xb     = (__bf16*)(ws + 0);            // 512 KB
    __bf16* h0b    = (__bf16*)(ws + 524288);       // h0[0] bf16
    __bf16* h1inb  = (__bf16*)(ws + 1048576);      // h0[1] bf16
    __bf16* h1b    = (__bf16*)(ws + 1572864);      // layer0 output bf16
    __bf16* h2b    = (__bf16*)(ws + 2097152);      // layer1 output bf16
    float*  gatesT = (float*) (ws + 2621440);      // [16384, 64] fp32, 4 MB (reused)
    float*  hidT   = (float*) (ws + 6815744);      // [2048, 64] fp32

    const int n4 = BH / 4;
    const dim3 cvtGrid((n4 + 255) / 256), blk(256);

    // 1) convert activations to bf16
    f32_to_bf16_kernel<<<cvtGrid, blk, 0, stream>>>(x,       xb,    n4);
    f32_to_bf16_kernel<<<cvtGrid, blk, 0, stream>>>(h0,      h0b,   n4);
    f32_to_bf16_kernel<<<cvtGrid, blk, 0, stream>>>(h0 + BH, h1inb, n4);

    // 2) layer 0: gates = W_ih0*x + W_hh0*h0[0] + biases  (N = 16384, K = 4096)
    gemm_gates_kernel<true, false><<<dim3(16384 / 128), blk, 0, stream>>>(
        W_ih0, W_hh0, xb, h0b, b_ih0, b_hh0, gatesT, H_DIM);
    lstm_pointwise_kernel<<<dim3(H_DIM / 64), blk, 0, stream>>>(
        gatesT, c0, out_h, out_c, nullptr, h1b);

    // 3) layer 1: gates = W_ih1*h1 + W_hh1*h0[1] + biases
    gemm_gates_kernel<true, false><<<dim3(16384 / 128), blk, 0, stream>>>(
        W_ih1, W_hh1, h1b, h1inb, b_ih1, b_hh1, gatesT, H_DIM);
    lstm_pointwise_kernel<<<dim3(H_DIM / 64), blk, 0, stream>>>(
        gatesT, c0 + BH, out_h + BH, out_c + BH, out_ms, h2b);

    // 4) count head: hid = relu(h2 @ Wc1^T + bc1)  (N = 2048, K = 4096)
    gemm_gates_kernel<false, true><<<dim3(2048 / 128), blk, 0, stream>>>(
        Wc1, nullptr, h2b, nullptr, bc1, nullptr, hidT, H_DIM);

    // 5) count = relu(hid @ Wc2^T + bc2)
    count_head_kernel<<<dim3(1), blk, 0, stream>>>(hidT, Wc2, bc2, out_cnt);
}
